// MultiheadAttention_10136122819096
// MI455X (gfx1250) — matrix operand order repaired and checked by reading, not compiled, not run on hardware
//
#include <hip/hip_runtime.h>
#include <hip/hip_bf16.h>

// ---------------------------------------------------------------------------
// MultiheadAttention forward for MI455X (gfx1250), wave32 + WMMA f16->f32.
// B=2, N=2048, D=1024, H=16, Hd=64.
// Compute-bound (~103 GFLOP vs ~80MB traffic @ 23.3TB/s) -> all matmuls on
// the WMMA pipe. Data movement uses the CDNA5-specific paths:
//   - GLOBAL_LOAD_ASYNC_TO_LDS_B128 (ASYNCcnt) with double-buffered LDS so
//     DMA of slab k+1 overlaps the WMMAs of slab k
//   - Tensor Data Mover (tensor_load_to_lds, TENSORcnt) for the one-shot Q
//     tile, using D# pad fields to insert the LDS row padding natively
//   - global_prefetch_b8 to warm L2 two slabs ahead
// ---------------------------------------------------------------------------

typedef __attribute__((ext_vector_type(16))) _Float16 v16h;
typedef __attribute__((ext_vector_type(8)))  float    v8f;
typedef __attribute__((ext_vector_type(4)))  unsigned v4u;
typedef __attribute__((ext_vector_type(8)))  unsigned v8u;

#define WMMA_F16(a, b, c) \
  __builtin_amdgcn_wmma_f32_16x16x32_f16(false, (a), false, (b), (short)0, (c), false, false)

constexpr int Bc   = 2;
constexpr int Nseq = 2048;
constexpr int Dm   = 1024;
constexpr int Hh   = 16;
constexpr int Hd   = 64;
constexpr int BN   = Bc * Nseq;   // 4096 rows total

static_assert(sizeof(_Float16) == 2, "f16 size");

// ---------------------------------------------------------------------------
// CDNA5 async DMA: GLOBAL_LOAD_ASYNC_TO_LDS_B128 (GVS mode: SGPR base + VGPR
// 32-bit byte offset; VDST VGPR carries the LDS byte address — low 32 bits of
// a flat shared pointer are the LDS offset per the aperture rules).
// ---------------------------------------------------------------------------
__device__ __forceinline__ void async_load_b128(const void* gbase_uniform,
                                                unsigned byte_off,
                                                void* lds_ptr) {
  unsigned lds = (unsigned)(uintptr_t)lds_ptr;
  asm volatile("global_load_async_to_lds_b128 %0, %1, %2"
               :
               : "v"(lds), "v"(byte_off),
                 "s"((unsigned long long)(uintptr_t)gbase_uniform)
               : "memory");
}

__device__ __forceinline__ void wait_async0() {
  asm volatile("s_wait_asynccnt 0x0" ::: "memory");
}

// ---------------------------------------------------------------------------
// LDS fragment loaders (wave32; contiguous k = 16*(lane/16) + i in element i of A and B)
// ---------------------------------------------------------------------------

// A-matrix 16x32 f16, source row-major (row stride ld halfs).
__device__ __forceinline__ v16h lds_a_frag(const _Float16* __restrict__ base, int ld) {
  const int lane = threadIdx.x & 31;
  const int row  = lane & 15;
  const int koff = (lane >> 4) << 4;           // 0 or 16
  const _Float16* p = base + row * ld;
  v16h a;
#pragma unroll
  for (int v = 0; v < 8; ++v) {
    const int k = 2 * v + koff;
    a[2 * v]     = p[k];
    a[2 * v + 1] = p[k + 1];
  }
  return a;
}

// B-matrix 32x16 where B(k,n) = M[n][k] (transposed source).
__device__ __forceinline__ v16h lds_bt_frag(const _Float16* __restrict__ base, int ld) {
  const int lane  = threadIdx.x & 31;
  const int n     = lane & 15;
  const int kbase = (lane >> 4) << 4;
  const _Float16* p = base + n * ld + kbase;
  v16h b;
#pragma unroll
  for (int i = 0; i < 16; ++i) b[i] = p[i];
  return b;
}

// B-matrix 32x16 where B(k,n) = M[k][n] (natural row-major source, e.g. V).
__device__ __forceinline__ v16h lds_b_frag(const _Float16* __restrict__ base, int ld) {
  const int lane  = threadIdx.x & 31;
  const int n     = lane & 15;
  const int kbase = (lane >> 4) << 4;
  v16h b;
#pragma unroll
  for (int i = 0; i < 16; ++i) b[i] = base[(kbase + i) * ld + n];
  return b;
}

// ---------------------------------------------------------------------------
// fp32 -> fp16 convert (grid-stride)
// ---------------------------------------------------------------------------
__global__ void cvt_f32_f16_kernel(const float* __restrict__ src,
                                   _Float16* __restrict__ dst, int n) {
  int i      = blockIdx.x * blockDim.x + threadIdx.x;
  int stride = gridDim.x * blockDim.x;
  for (; i < n; i += stride) dst[i] = (_Float16)src[i];
}

// ---------------------------------------------------------------------------
// GEMM: out[r, j] = sum_d A[r, d] * W[j, d] + bias[j]
// Block: 256 threads (8 waves), 128x64 output tile; each wave owns 16 rows x
// 64 cols (4 accumulators). 64-wide K slabs, double-buffered async DMA:
// slab k+1 streams in while slab k's 8 WMMAs run.
// mode 0: f16 out in (B,H,N,Hd).  mode 1: f32 out row-major (final proj).
// ---------------------------------------------------------------------------
__global__ __launch_bounds__(256)
void gemm_xwt_kernel(const _Float16* __restrict__ A,
                     const _Float16* __restrict__ W,
                     const float* __restrict__ bias,
                     _Float16* __restrict__ out_h,
                     float* __restrict__ out_f,
                     int mode) {
  __shared__ _Float16 Ash[2][128 * 72];   // 128 rows x 64 K (stride 72)
  __shared__ _Float16 Wsh[2][64 * 72];    // 64 rows  x 64 K

  const int m0   = blockIdx.x * 128;
  const int j0   = blockIdx.y * 64;
  const int t    = threadIdx.x;
  const int wave = t >> 5;
  const int lane = t & 31;
  const int half = lane >> 4;
  const int nloc = lane & 15;

  const _Float16* Abase = A + (size_t)m0 * Dm;
  const _Float16* Wbase = W + (size_t)j0 * Dm;

  auto stage = [&](int buf, int k0) {
#pragma unroll
    for (int i = 0; i < 4; ++i) {          // A: 1024 b128 chunks, 4/thread
      const int chunk = t * 4 + i;
      const int r = chunk >> 3;
      const int c = (chunk & 7) * 8;
      async_load_b128(Abase, (unsigned)((r * Dm + k0 + c) * 2),
                      &Ash[buf][r * 72 + c]);
    }
#pragma unroll
    for (int i = 0; i < 2; ++i) {          // W: 512 b128 chunks, 2/thread
      const int chunk = t * 2 + i;
      const int r = chunk >> 3;
      const int c = (chunk & 7) * 8;
      async_load_b128(Wbase, (unsigned)((r * Dm + k0 + c) * 2),
                      &Wsh[buf][r * 72 + c]);
    }
  };

  v8f acc[4] = {{}, {}, {}, {}};

  stage(0, 0);
  wait_async0();
  __syncthreads();

  int cur = 0;
  for (int k0 = 0; k0 < Dm; k0 += 64) {
    if (k0 + 64 < Dm) {
      stage(cur ^ 1, k0 + 64);             // DMA next slab, overlapped
      if (k0 + 128 < Dm) {                 // warm L2 two slabs ahead
        __builtin_prefetch(Abase + (size_t)(t >> 1) * Dm + k0 + 128 + (t & 1) * 32, 0, 0);
        __builtin_prefetch(Wbase + (size_t)(t >> 2) * Dm + k0 + 128 + (t & 3) * 16, 0, 0);
      }
    }

    // ---- 8 WMMAs per slab: 2 K-steps x 4 column tiles
#pragma unroll
    for (int kk = 0; kk < 64; kk += 32) {
      v16h a = lds_a_frag(&Ash[cur][(wave * 16) * 72 + kk], 72);
#pragma unroll
      for (int nt = 0; nt < 4; ++nt) {
        v16h b = lds_bt_frag(&Wsh[cur][(nt * 16) * 72 + kk], 72);
        acc[nt] = WMMA_F16(a, b, acc[nt]);
      }
    }

    wait_async0();       // own DMAs done ...
    __syncthreads();     // ... then everyone's are
    cur ^= 1;
  }

  // ---- epilogue: C/D layout (lane: col = lane&15, rows v + 8*(lane>>4))
#pragma unroll
  for (int nt = 0; nt < 4; ++nt) {
    const int col = j0 + nt * 16 + nloc;
    const float bv = bias[col];
    if (mode == 0) {
      const int h  = col >> 6;
      const int hd = col & 63;
#pragma unroll
      for (int v = 0; v < 8; ++v) {
        const int r  = m0 + wave * 16 + v + 8 * half;
        const int b  = r >> 11;                 // / Nseq
        const int ns = r & (Nseq - 1);
        out_h[(((size_t)(b * Hh + h) * Nseq) + ns) * Hd + hd] =
            (_Float16)(acc[nt][v] + bv);
      }
    } else {
#pragma unroll
      for (int v = 0; v < 8; ++v) {
        const int r = m0 + wave * 16 + v + 8 * half;
        out_f[(size_t)r * Dm + col] = acc[nt][v] + bv;
      }
    }
  }
}

// ---------------------------------------------------------------------------
// Flash attention: one block per (b*H + h, 128 query rows); 8 waves, each wave
// owns 16 query rows x full Hd=64. Q tile loaded by the Tensor Data Mover
// (padded 2D tile), K/V tiles double-buffered via async DMA.
// ---------------------------------------------------------------------------
__global__ __launch_bounds__(256)
void flash_attn_kernel(const _Float16* __restrict__ Qh,
                       const _Float16* __restrict__ Kh,
                       const _Float16* __restrict__ Vh,
                       _Float16* __restrict__ Oh,
                       const int* __restrict__ causal_flag) {
  __shared__ _Float16 Qs[128 * 72];       // 128 q-rows x 64 hd (stride 72)
  __shared__ _Float16 Ks[2][64 * 72];     // 64 keys x 64 hd, double-buffered
  __shared__ _Float16 Vs[2][64 * 72];
  __shared__ _Float16 Ps[8 * 16 * 72];    // per-wave P scratch (16 x 64 keys)

  const int bh   = blockIdx.x;            // b*H + h
  const int q0   = blockIdx.y * 128;
  const int t    = threadIdx.x;
  const int wave = t >> 5;
  const int lane = t & 31;
  const int half = lane >> 4;
  const int nloc = lane & 15;
  const bool causal = (*causal_flag) != 0;

  const _Float16* Qbase = Qh + ((size_t)bh * Nseq + q0) * Hd;
  const _Float16* Kbase = Kh + (size_t)bh * Nseq * Hd;
  const _Float16* Vbase = Vh + (size_t)bh * Nseq * Hd;

  // ---- Q tile via Tensor Data Mover: 2D tile, dim0=64 elems x dim1=128 rows,
  // data_size=2B. pad_interval=32 DWORDs (one 128B row), pad_amount=4 DWORDs
  // (16B) -> LDS row stride 72 halfs, matching the fragment loaders.
  if (wave == 0) {
    const unsigned lds = (unsigned)(uintptr_t)(void*)Qs;
    const unsigned long long ga = (unsigned long long)(uintptr_t)Qbase;
    v4u g0;
    g0[0] = 1u;                                          // count=1, load
    g0[1] = lds;                                         // lds_addr
    g0[2] = (unsigned)ga;                                // global_addr[31:0]
    g0[3] = (unsigned)((ga >> 32) & 0x01FFFFFFu)         // global_addr[56:32]
          | (2u << 30);                                  // type=2 (image)
    v8u g1;
    g1[0] = (1u << 16)        // data_size = 2 bytes
          | (1u << 20)        // pad_enable
          | (4u << 22)        // pad_interval: 32 DWORDs
          | (3u << 25);       // pad_amount: 4 DWORDs
    g1[1] = (64u & 0xFFFFu) << 16;   // tensor_dim0 = 64 (bits 48..63)
    g1[2] = (128u << 16);            // tensor_dim0 hi | tensor_dim1 = 128 lo
    g1[3] = (64u << 16);             // tensor_dim1 hi | tile_dim0 = 64
    g1[4] = 128u;                    // tile_dim1 = 128, tile_dim2 = 0
    g1[5] = 64u;                     // tensor_dim0_stride = 64 (lo)
    g1[6] = 0u;                      // stride hi | dim1_stride lo
    g1[7] = 0u;
    asm volatile("tensor_load_to_lds %0, %1" :: "s"(g0), "s"(g1) : "memory");
    __builtin_amdgcn_s_wait_tensorcnt(0);
  }

  auto stage_kv = [&](int buf, int kt2) {
    const int r     = t >> 2;
    const int cbase = (t & 3) * 16;
#pragma unroll
    for (int i = 0; i < 2; ++i) {
      const int c = cbase + i * 8;
      const unsigned off = (unsigned)(((kt2 * 64 + r) * Hd + c) * 2);
      async_load_b128(Kbase, off, &Ks[buf][r * 72 + c]);
      async_load_b128(Vbase, off, &Vs[buf][r * 72 + c]);
    }
  };

  float row_m[8], row_l[8];
#pragma unroll
  for (int v = 0; v < 8; ++v) { row_m[v] = -1e30f; row_l[v] = 0.0f; }
  v8f acc[4] = {{}, {}, {}, {}};

  const int NT = Nseq / 64;   // 32 key tiles
  int nkt = NT;
  if (causal) {
    int lim = (q0 + 127) / 64 + 1;
    nkt = lim < NT ? lim : NT;
  }

  stage_kv(0, 0);
  wait_async0();
  __syncthreads();            // Q (TDM) + first K/V visible to all waves

  int cur = 0;
  for (int kt = 0; kt < nkt; ++kt) {
    if (kt + 1 < nkt) {
      stage_kv(cur ^ 1, kt + 1);     // DMA next key tile, overlapped
      if (kt + 2 < nkt) {
        const size_t p = (size_t)((kt + 2) * 64 + (t >> 2)) * Hd + (t & 3) * 16;
        __builtin_prefetch(Kbase + p, 0, 0);
        __builtin_prefetch(Vbase + p, 0, 0);
      }
    }

    // ---- S = Q * K^T : wave computes 16 x 64 in four 16x16 tiles ----
    v8f s[4] = {{}, {}, {}, {}};
#pragma unroll
    for (int kk = 0; kk < Hd; kk += 32) {
      v16h a = lds_a_frag(Qs + (wave * 16) * 72 + kk, 72);
#pragma unroll
      for (int nt = 0; nt < 4; ++nt) {
        v16h b = lds_bt_frag(&Ks[cur][(nt * 16) * 72 + kk], 72);
        s[nt] = WMMA_F16(a, b, s[nt]);
      }
    }

    // ---- scale + causal mask + per-row max over this tile ----
    float tmax[8];
#pragma unroll
    for (int v = 0; v < 8; ++v) tmax[v] = -1e30f;
#pragma unroll
    for (int nt = 0; nt < 4; ++nt) {
      const int key = kt * 64 + nt * 16 + nloc;
#pragma unroll
      for (int v = 0; v < 8; ++v) {
        float x = s[nt][v] * 0.125f;   // 1/sqrt(64)
        if (causal) {
          const int q = q0 + wave * 16 + v + 8 * half;
          if (key > q) x = -1e30f;
        }
        s[nt][v] = x;
        tmax[v]  = fmaxf(tmax[v], x);
      }
    }
    // reduce across the 16 lanes sharing each row (stays within half-wave)
#pragma unroll
    for (int off = 8; off >= 1; off >>= 1) {
#pragma unroll
      for (int v = 0; v < 8; ++v)
        tmax[v] = fmaxf(tmax[v], __shfl_xor(tmax[v], off, 32));
    }

    float rescale[8], tsum[8];
#pragma unroll
    for (int v = 0; v < 8; ++v) {
      const float nm = fmaxf(row_m[v], tmax[v]);
      rescale[v] = __expf(row_m[v] - nm);
      row_m[v]   = nm;
      tsum[v]    = 0.0f;
    }

    // ---- P = exp(s - m), accumulate row sums, spill P to LDS (A-layout src)
#pragma unroll
    for (int nt = 0; nt < 4; ++nt) {
#pragma unroll
      for (int v = 0; v < 8; ++v) {
        const float p = __expf(s[nt][v] - row_m[v]);
        tsum[v] += p;
        Ps[(wave * 16 + v + 8 * half) * 72 + nt * 16 + nloc] = (_Float16)p;
      }
    }
#pragma unroll
    for (int off = 8; off >= 1; off >>= 1) {
#pragma unroll
      for (int v = 0; v < 8; ++v)
        tsum[v] += __shfl_xor(tsum[v], off, 32);
    }
#pragma unroll
    for (int v = 0; v < 8; ++v) {
      row_l[v] = row_l[v] * rescale[v] + tsum[v];
#pragma unroll
      for (int j = 0; j < 4; ++j) acc[j][v] *= rescale[v];
    }

    // ---- O += P * V (same-wave DS ops are in-order; Ps is wave-private) ----
#pragma unroll
    for (int kk = 0; kk < 64; kk += 32) {
      v16h a = lds_a_frag(Ps + (wave * 16) * 72 + kk, 72);
#pragma unroll
      for (int j = 0; j < 4; ++j) {
        v16h b = lds_b_frag(&Vs[cur][kk * 72 + j * 16], 72);
        acc[j] = WMMA_F16(a, b, acc[j]);
      }
    }

    wait_async0();     // own next-tile DMAs done ...
    __syncthreads();   // ... then everyone's; safe to swap buffers
    cur ^= 1;
  }

  // ---- epilogue: O /= l, write to (B, N, D) f16 for the final projection
  {
    const int b = bh >> 4;      // / Hh
    const int h = bh & 15;
#pragma unroll
    for (int v = 0; v < 8; ++v) {
      const float inv = 1.0f / row_l[v];
      const int r = q0 + wave * 16 + v + 8 * half;
#pragma unroll
      for (int j = 0; j < 4; ++j) {
        Oh[((size_t)b * Nseq + r) * Dm + h * 64 + j * 16 + nloc] =
            (_Float16)(acc[j][v] * inv);
      }
    }
  }
}

// ---------------------------------------------------------------------------
// Host-side launch
// ---------------------------------------------------------------------------
extern "C" void kernel_launch(void* const* d_in, const int* in_sizes, int n_in,
                              void* d_out, int out_size, void* d_ws, size_t ws_size,
                              hipStream_t stream) {
  const float* x  = (const float*)d_in[0];
  const float* Wq = (const float*)d_in[1];
  const float* bq = (const float*)d_in[2];
  const float* Wk = (const float*)d_in[3];
  const float* bk = (const float*)d_in[4];
  const float* Wv = (const float*)d_in[5];
  const float* bv = (const float*)d_in[6];
  const float* Wo = (const float*)d_in[7];
  const float* bo = (const float*)d_in[8];
  const int* is_causal = (const int*)d_in[9];
  float* out = (float*)d_out;

  char* ws = (char*)d_ws;
  const size_t MB = 1u << 20;
  _Float16* xh  = (_Float16*)(ws);             // 8 MB  (BN x D)
  _Float16* Wqh = (_Float16*)(ws + 8  * MB);   // 2 MB
  _Float16* Wkh = (_Float16*)(ws + 10 * MB);   // 2 MB
  _Float16* Wvh = (_Float16*)(ws + 12 * MB);   // 2 MB
  _Float16* Woh = (_Float16*)(ws + 14 * MB);   // 2 MB
  _Float16* Qh  = (_Float16*)(ws + 16 * MB);   // 8 MB  (B,H,N,Hd)
  _Float16* Kh  = (_Float16*)(ws + 24 * MB);   // 8 MB
  _Float16* Vh  = (_Float16*)(ws + 32 * MB);   // 8 MB
  _Float16* Oh  = (_Float16*)(ws + 40 * MB);   // 8 MB  (B,N,D)

  const int nx = BN * Dm;       // 4194304
  const int nw = Dm * Dm;       // 1048576
  cvt_f32_f16_kernel<<<1024, 256, 0, stream>>>(x,  xh,  nx);
  cvt_f32_f16_kernel<<<512,  256, 0, stream>>>(Wq, Wqh, nw);
  cvt_f32_f16_kernel<<<512,  256, 0, stream>>>(Wk, Wkh, nw);
  cvt_f32_f16_kernel<<<512,  256, 0, stream>>>(Wv, Wvh, nw);
  cvt_f32_f16_kernel<<<512,  256, 0, stream>>>(Wo, Woh, nw);

  dim3 ggemm(BN / 128, Dm / 64);   // 32 x 16
  gemm_xwt_kernel<<<ggemm, 256, 0, stream>>>(xh, Wqh, bq, Qh, nullptr, 0);
  gemm_xwt_kernel<<<ggemm, 256, 0, stream>>>(xh, Wkh, bk, Kh, nullptr, 0);
  gemm_xwt_kernel<<<ggemm, 256, 0, stream>>>(xh, Wvh, bv, Vh, nullptr, 0);

  dim3 gattn(Bc * Hh, Nseq / 128);  // 32 x 16
  flash_attn_kernel<<<gattn, 256, 0, stream>>>(Qh, Kh, Vh, Oh, is_causal);

  gemm_xwt_kernel<<<ggemm, 256, 0, stream>>>(Oh, Woh, bo, nullptr, out, 1);
}
